// SNN_59390807769670
// MI455X (gfx1250) — matrix operand order repaired and checked by reading, not compiled, not run on hardware
//
#include <hip/hip_runtime.h>
#include <stdint.h>

#define T_STEPS 25
#define ROWD    1024
#define MDIM    (T_STEPS * ROWD)   // 25600
#define KDIM    784
#define KP      800                // padded K: 25 * 32, no tail handling
#define HDIM    1000
#define HP      1024               // padded H: 16 * 64, no N guards on loads
#define ODIM    10
#define BETA    0.95f

typedef __attribute__((ext_vector_type(16))) __bf16 v16bf;
typedef __attribute__((ext_vector_type(8)))  float  v8f;

union BFrag { v16bf v; unsigned int u[8]; uint4 q[2]; };

// round-to-nearest-even f32 -> bf16, packed pair (lo = even-k element)
__device__ __forceinline__ unsigned int pk_bf16(float a, float b) {
    unsigned int ua = __float_as_uint(a);
    unsigned int ub = __float_as_uint(b);
    ua += 0x7FFFu + ((ua >> 16) & 1u);
    ub += 0x7FFFu + ((ub >> 16) & 1u);
    return (ua >> 16) | (ub & 0xFFFF0000u);
}

// ---------------------------------------------------------------------------
// Pre-pass A: data (25600x784 f32) -> dataBF (25600x800 bf16, zero-padded K)
// one thread per bf16 pair
// ---------------------------------------------------------------------------
__global__ void __launch_bounds__(256)
cvtA_kernel(const float* __restrict__ src, unsigned short* __restrict__ dst) {
    int idx = blockIdx.x * blockDim.x + threadIdx.x;
    if (idx >= MDIM * (KP / 2)) return;
    int m  = idx / (KP / 2);
    int k2 = (idx - m * (KP / 2)) * 2;
    unsigned int packed = 0u;
    if (k2 < KDIM) {                               // KDIM even: pairs never straddle
        const float* p = src + (size_t)m * KDIM + k2;
        packed = pk_bf16(p[0], p[1]);
    }
    *(unsigned int*)(dst + (size_t)m * KP + k2) = packed;
}

// ---------------------------------------------------------------------------
// Pre-pass B: w1 (1000x784 f32) -> w1BF (1024x800 bf16, zero-padded K and H)
// ---------------------------------------------------------------------------
__global__ void __launch_bounds__(256)
cvtB_kernel(const float* __restrict__ src, unsigned short* __restrict__ dst) {
    int idx = blockIdx.x * blockDim.x + threadIdx.x;
    if (idx >= HP * (KP / 2)) return;
    int n  = idx / (KP / 2);
    int k2 = (idx - n * (KP / 2)) * 2;
    unsigned int packed = 0u;
    if (n < HDIM && k2 < KDIM) {
        const float* p = src + (size_t)n * KDIM + k2;
        packed = pk_bf16(p[0], p[1]);
    }
    *(unsigned int*)(dst + (size_t)n * KP + k2) = packed;
}

// ---------------------------------------------------------------------------
// Phase 1: cur1[m,h] = sum_k data[m,k]*w1[h,k] + b1[h]
// One wave computes 32(M) x 64(N): 2 A frags x 4 B frags = 8 WMMA per K-step.
// Inputs pre-converted/padded: zero guards, zero conversion in the loop.
// ---------------------------------------------------------------------------
__global__ void __launch_bounds__(128)
gemm1_wmma(const unsigned short* __restrict__ A16,
           const unsigned short* __restrict__ B16,
           const float* __restrict__ b1, float* __restrict__ cur1) {
    const int wave = blockIdx.x * (blockDim.x >> 5) + (threadIdx.x >> 5);
    const int lane = threadIdx.x & 31;
    const int lrow = lane & 15;
    const int half = lane >> 4;

    const int NT    = 16;              // 1024 / 64
    const int mtile = wave / NT;       // 0..799, 32 rows each
    const int ntile = wave % NT;       // 0..15,  64 cols each
    const int M0 = mtile * 32;
    const int N0 = ntile * 64;

    const unsigned short* pa0 = A16 + (size_t)(M0 + lrow) * KP;
    const unsigned short* pa1 = A16 + (size_t)(M0 + 16 + lrow) * KP;
    const unsigned short* pbs[4];
#pragma unroll
    for (int s = 0; s < 4; ++s)
        pbs[s] = B16 + (size_t)(N0 + s * 16 + lrow) * KP;

    v8f acc0[4], acc1[4];
#pragma unroll
    for (int s = 0; s < 4; ++s)
#pragma unroll
        for (int i = 0; i < 8; ++i) { acc0[s][i] = 0.0f; acc1[s][i] = 0.0f; }

    const int aoff = half * 8;         // A: lanes>=16 hold K sub-range +8
    const int boff = half * 8;         // B: lanes>=16 hold K sub-range +8 (same as A)

    for (int k0 = 0; k0 < KP; k0 += 32) {
        // next-iteration prefetch (global_prefetch_b8)
        if (k0 + 32 < KP) {
            __builtin_prefetch(pa0 + k0 + 32 + aoff, 0, 3);
            __builtin_prefetch(pa1 + k0 + 32 + aoff, 0, 3);
        }
        BFrag A0, A1;
        A0.q[0] = *(const uint4*)(pa0 + k0 + aoff);        // K = k0+half*8 .. +7
        A0.q[1] = *(const uint4*)(pa0 + k0 + 16 + aoff);   // K = k0+16+half*8 .. +7
        A1.q[0] = *(const uint4*)(pa1 + k0 + aoff);
        A1.q[1] = *(const uint4*)(pa1 + k0 + 16 + aoff);
#pragma unroll
        for (int s = 0; s < 4; ++s) {
            BFrag B;
            B.q[0] = *(const uint4*)(pbs[s] + k0 + boff);
            B.q[1] = *(const uint4*)(pbs[s] + k0 + 16 + boff);
            acc0[s] = __builtin_amdgcn_wmma_f32_16x16x32_bf16(
                false, A0.v, false, B.v, (short)0, acc0[s], false, false);
            acc1[s] = __builtin_amdgcn_wmma_f32_16x16x32_bf16(
                false, A1.v, false, B.v, (short)0, acc1[s], false, false);
        }
    }

    // store C + bias: VGPR r of a frag holds row (base + r + 8*half), col fixed/lane
#pragma unroll
    for (int s = 0; s < 4; ++s) {
        int n = N0 + s * 16 + lrow;
        if (n >= HDIM) continue;
        const float bias = b1[n];
        float* pc0 = cur1 + (size_t)(M0 + 8 * half) * HDIM + n;
        float* pc1 = cur1 + (size_t)(M0 + 16 + 8 * half) * HDIM + n;
#pragma unroll
        for (int r = 0; r < 8; ++r) {
            pc0[(size_t)r * HDIM] = acc0[s][r] + bias;
            pc1[(size_t)r * HDIM] = acc1[s][r] + bias;
        }
    }
}

// ---------------------------------------------------------------------------
// Phase 2: LIF layer 1. 1000 independent chains of 25600 steps (in-place).
// ---------------------------------------------------------------------------
__global__ void __launch_bounds__(256)
lif1_kernel(float* __restrict__ cur1) {
    int h = blockIdx.x * blockDim.x + threadIdx.x;
    if (h >= HDIM) return;
    float* p  = cur1 + h;
    float mem = 0.0f;
    for (int t0 = 0; t0 < MDIM; t0 += 8) {
        float c[8];
#pragma unroll
        for (int u = 0; u < 8; ++u) c[u] = p[(size_t)(t0 + u) * HDIM];
        float s[8];
#pragma unroll
        for (int u = 0; u < 8; ++u) {
            mem = fmaf(BETA, mem, c[u]);
            s[u] = (mem >= 1.0f) ? 1.0f : 0.0f;
            mem -= s[u];
        }
#pragma unroll
        for (int u = 0; u < 8; ++u) p[(size_t)(t0 + u) * HDIM] = s[u];
    }
}

// ---------------------------------------------------------------------------
// Phase 3: cur2[t,o] = spk1[t,:] . w2[o,:] + b2[o]   (o padded to 16)
// ---------------------------------------------------------------------------
__global__ void __launch_bounds__(256)
fc2_kernel(const float* __restrict__ spk1, const float* __restrict__ w2,
           const float* __restrict__ b2, float* __restrict__ cur2) {
    int gid = blockIdx.x * blockDim.x + threadIdx.x;
    int t = gid >> 4;
    int o = gid & 15;
    if (t >= MDIM) return;
    float sum = 0.0f;
    if (o < ODIM) {
        const float4* sp = (const float4*)(spk1 + (size_t)t * HDIM);
        const float4* wr = (const float4*)(w2 + (size_t)o * HDIM);
#pragma unroll 5
        for (int k = 0; k < HDIM / 4; ++k) {
            float4 a = sp[k], b = wr[k];
            sum = fmaf(a.x, b.x, sum); sum = fmaf(a.y, b.y, sum);
            sum = fmaf(a.z, b.z, sum); sum = fmaf(a.w, b.w, sum);
        }
        sum += b2[o];
    }
    cur2[(size_t)t * 16 + o] = sum;
}

// ---------------------------------------------------------------------------
// Phase 4: LIF layer 2. 10 chains of 25600 steps, single wave.
// spk2 written to both output halves (reference: mem2_rec == spk2_rec).
// ---------------------------------------------------------------------------
__global__ void
lif2_kernel(const float* __restrict__ cur2, float* __restrict__ out) {
    int o = threadIdx.x;
    if (o >= 16) return;
    const bool act = (o < ODIM);
    float* out_spk = out;
    float* out_mem = out + (size_t)MDIM * ODIM;
    float  mem = 0.0f;
    for (int t0 = 0; t0 < MDIM; t0 += 8) {
        float c[8];
#pragma unroll
        for (int u = 0; u < 8; ++u) c[u] = cur2[(size_t)(t0 + u) * 16 + o];
#pragma unroll
        for (int u = 0; u < 8; ++u) {
            mem = fmaf(BETA, mem, c[u]);
            float s = (mem >= 1.0f) ? 1.0f : 0.0f;
            mem -= s;
            if (act) {
                int idx = (t0 + u) * ODIM + o;
                out_spk[idx] = s;
                out_mem[idx] = s;
            }
        }
    }
}

extern "C" void kernel_launch(void* const* d_in, const int* in_sizes, int n_in,
                              void* d_out, int out_size, void* d_ws, size_t ws_size,
                              hipStream_t stream) {
    const float* data = (const float*)d_in[0];  // (25,1024,784)
    const float* w1   = (const float*)d_in[1];  // (1000,784)
    const float* b1   = (const float*)d_in[2];  // (1000,)
    const float* w2   = (const float*)d_in[3];  // (10,1000)
    const float* b2   = (const float*)d_in[4];  // (10,)
    float* out = (float*)d_out;                 // 2 * 25600*10

    // workspace layout (all 256B-aligned)
    char* ws = (char*)d_ws;
    float*          cur1   = (float*)ws;                                    // 102,400,000 B
    unsigned short* dataBF = (unsigned short*)(ws + 102400000);             //  40,960,000 B
    unsigned short* w1BF   = (unsigned short*)(ws + 102400000 + 40960000);  //   1,638,400 B
    float*          cur2   = (float*)(ws + 102400000 + 40960000 + 1638400); //   1,638,400 B

    // pre-convert + pad to bf16 (eliminates all guards/conversion in the GEMM loop)
    cvtA_kernel<<<(MDIM * (KP / 2)) / 256, 256, 0, stream>>>(data, dataBF);
    cvtB_kernel<<<(HP * (KP / 2)) / 256, 256, 0, stream>>>(w1, w1BF);

    // GEMM1: 800 M32-tiles x 16 N64-tiles = 12800 waves, 4 waves/block
    gemm1_wmma<<<3200, 128, 0, stream>>>(dataBF, w1BF, b1, cur1);
    // LIF1: 1000 neuron chains (in-place spikes)
    lif1_kernel<<<4, 256, 0, stream>>>(cur1);
    // FC2: 25600*16 threads
    fc2_kernel<<<(MDIM * 16) / 256, 256, 0, stream>>>(cur1, w2, b2, cur2);
    // LIF2: one wave
    lif2_kernel<<<1, 32, 0, stream>>>(cur2, out);
}
